// GraphSAGEFraud_73212012527801
// MI455X (gfx1250) — matrix operand order repaired and checked by reading, not compiled, not run on hardware
//
#include <hip/hip_runtime.h>
#include <stdint.h>

#ifndef USE_ASYNC_LDS
#define USE_ASYNC_LDS 1
#endif

// ---------- bf16 helpers ----------
__device__ __forceinline__ unsigned short f32_to_bf16(float f) {
  unsigned int u = __float_as_uint(f);
  u += 0x7FFFu + ((u >> 16) & 1u);   // round-to-nearest-even
  return (unsigned short)(u >> 16);
}

// ---------- utility kernels ----------
__global__ void fill_zero_f32(float* __restrict__ p, int n) {
  int i = blockIdx.x * blockDim.x + threadIdx.x;
  if (i < n) p[i] = 0.0f;
}

__global__ void count_edges(const int* __restrict__ dst, int E, float* __restrict__ cnt) {
  int e = blockIdx.x * blockDim.x + threadIdx.x;
  if (e < E) atomicAdd(&cnt[dst[e]], 1.0f);
}

__global__ void invert_cnt(float* __restrict__ cnt, int n) {
  int i = blockIdx.x * blockDim.x + threadIdx.x;
  if (i < n) cnt[i] = 1.0f / fmaxf(cnt[i], 1.0f);
}

// One float4 of features per thread; d/4 threads per edge.
__global__ void scatter_edges(const float* __restrict__ x, const int* __restrict__ src,
                              const int* __restrict__ dst, float* __restrict__ agg,
                              int E, int d) {
  int gid = blockIdx.x * blockDim.x + threadIdx.x;
  int groups = d >> 2;
  int e = gid / groups;
  if (e >= E) return;
  int fg = gid - e * groups;
  int s = src[e], t = dst[e];
  float4 v = reinterpret_cast<const float4*>(x + (size_t)s * d)[fg];
  float* a = agg + (size_t)t * d + (size_t)fg * 4;
  atomicAdd(a + 0, v.x);
  atomicAdd(a + 1, v.y);
  atomicAdd(a + 2, v.z);
  atomicAdd(a + 3, v.w);
}

// Build A' = [mean | x] as packed bf16 pairs (one dword per thread).
// Row stride in dwords = din (== K2/2).
__global__ void build_abf(const float* __restrict__ agg, const float* __restrict__ xin,
                          const float* __restrict__ cinv, unsigned int* __restrict__ abf,
                          int n, int din) {
  int gid = blockIdx.x * blockDim.x + threadIdx.x;   // n * din pair-slots
  int node = gid / din;
  if (node >= n) return;
  int p = gid - node * din;
  int k0 = p * 2;
  float v0, v1;
  if (k0 < din) {
    float ci = cinv[node];
    v0 = agg[(size_t)node * din + k0] * ci;
    v1 = agg[(size_t)node * din + k0 + 1] * ci;
  } else {
    v0 = xin[(size_t)node * din + (k0 - din)];
    v1 = xin[(size_t)node * din + (k0 - din) + 1];
  }
  abf[(size_t)node * din + p] =
      (unsigned int)f32_to_bf16(v0) | ((unsigned int)f32_to_bf16(v1) << 16);
}

// Pack W' = [Wl | Wr] row-major bf16, shape [dout][2*din].
__global__ void build_wbf(const float* __restrict__ Wl, const float* __restrict__ Wr,
                          unsigned short* __restrict__ wbf, int dout, int din) {
  int gid = blockIdx.x * blockDim.x + threadIdx.x;
  int K2 = 2 * din;
  int j = gid / K2;
  if (j >= dout) return;
  int k = gid - j * K2;
  float v = (k < din) ? Wl[(size_t)j * din + k] : Wr[(size_t)j * din + (k - din)];
  wbf[(size_t)j * K2 + k] = f32_to_bf16(v);
}

// ---------- WMMA GEMM: out = relu(A' @ W'^T + b) ----------
typedef __attribute__((ext_vector_type(16))) __bf16 v16bf;
typedef __attribute__((ext_vector_type(8)))  float v8f;

union BFrag { unsigned int u[8]; v16bf v; };

// LDS A panel: 32 rows x kpairs(<=256) dwords; row stride padded to 260 dwords
// (260*4 = 1040 bytes, 16B-aligned so B128 async writes stay aligned).
#define LDS_STRIDE 260

// block = 256 threads (8 wave32s); block tile = 32 rows x 128 cols;
// each wave: two 16x16 sub-tiles sharing one B fragment.
__global__ void __launch_bounds__(256)
sage_gemm_wmma(const unsigned int* __restrict__ abf,   // [n][K2/2] dwords (bf16 pairs)
               const unsigned int* __restrict__ wbf,   // [OUT][K2/2] dwords (bf16 pairs)
               const float* __restrict__ bias,
               float* __restrict__ out,
               int n, int K2, int OUT) {
  __shared__ unsigned int Alds[32 * LDS_STRIDE];
  const int tid  = threadIdx.x;
  const int wave = tid >> 5;
  const int lane = tid & 31;
  const int half = lane >> 4;             // lane half selects K sub-range
  const int m    = lane & 15;             // A row / B-D column within tile
  const int rowBase = blockIdx.x * 32;
  const int col = blockIdx.y * 128 + wave * 16 + m;
  const int kpairs = K2 >> 1;             // 128 or 256
  const int chunksPerRow = kpairs >> 2;   // uint4 chunks per row

  // ---- Stage entire 32-row A panel into LDS once ----
  if (rowBase + 32 <= n) {
    const int totalChunks = 32 * chunksPerRow;   // multiple of 256
    for (int c = tid; c < totalChunks; c += 256) {
      int r  = c / chunksPerRow;
      int cp = c - r * chunksPerRow;
      const unsigned int* g = abf + (size_t)(rowBase + r) * kpairs + cp * 4;
#if USE_ASYNC_LDS
      unsigned int ldsoff = (unsigned int)((r * LDS_STRIDE + cp * 4) * 4); // bytes
      asm volatile("global_load_async_to_lds_b128 %0, %1, off"
                   :: "v"(ldsoff), "v"(g) : "memory");
#else
      *reinterpret_cast<uint4*>(&Alds[r * LDS_STRIDE + cp * 4]) =
          *reinterpret_cast<const uint4*>(g);
#endif
    }
#if USE_ASYNC_LDS
    asm volatile("s_wait_asynccnt 0x0" ::: "memory");
#endif
  } else {
    // Partial tile (never hit for N=100000, kept for generality): zero-fill.
    for (int c = tid; c < 32 * kpairs; c += 256) {
      int r = c / kpairs;
      int p = c - r * kpairs;
      int node = rowBase + r;
      Alds[r * LDS_STRIDE + p] = (node < n) ? abf[(size_t)node * kpairs + p] : 0u;
    }
  }
  __syncthreads();

  v8f acc0 = {};
  v8f acc1 = {};

  for (int kc = 0; kc < kpairs; kc += 16) {
    BFrag B;
#pragma unroll
    for (int j = 0; j < 8; ++j) {
      // B 32x16 bf16 in this code's contiguous k order: lanes 0-15 hold K 0-15, lanes 16-31 hold K 16-31
      B.u[j] = wbf[(size_t)col * kpairs + kc + 8 * half + j];
    }
    BFrag A0, A1;
#pragma unroll
    for (int j = 0; j < 8; ++j) {
      // A 16x32 bf16 in this code's contiguous k order: lanes 0-15 hold K 0-15, lanes 16-31 hold K 16-31
      int pa = kc + 8 * half + j;
      A0.u[j] = Alds[m * LDS_STRIDE + pa];
      A1.u[j] = Alds[(16 + m) * LDS_STRIDE + pa];
    }
    acc0 = __builtin_amdgcn_wmma_f32_16x16x32_bf16(false, A0.v, false, B.v,
                                                   (short)0, acc0, false, false);
    acc1 = __builtin_amdgcn_wmma_f32_16x16x32_bf16(false, A1.v, false, B.v,
                                                   (short)0, acc1, false, false);
  }

  float bj = bias[col];
#pragma unroll
  for (int r = 0; r < 8; ++r) {
    // D layout: M = r + 8*half, N = lane&15
    int row0 = rowBase + r + 8 * half;
    int row1 = row0 + 16;
    if (row0 < n) out[(size_t)row0 * OUT + col] = fmaxf(acc0[r] + bj, 0.0f);
    if (row1 < n) out[(size_t)row1 * OUT + col] = fmaxf(acc1[r] + bj, 0.0f);
  }
}

// ---------- host ----------
extern "C" void kernel_launch(void* const* d_in, const int* in_sizes, int n_in,
                              void* d_out, int out_size, void* d_ws, size_t ws_size,
                              hipStream_t stream) {
  const float* x   = (const float*)d_in[0];
  const int*   ei  = (const int*)d_in[1];
  const float* W1l = (const float*)d_in[2];
  const float* b1  = (const float*)d_in[3];
  const float* W1r = (const float*)d_in[4];
  const float* W2l = (const float*)d_in[5];
  const float* b2  = (const float*)d_in[6];
  const float* W2r = (const float*)d_in[7];
  const float* W3l = (const float*)d_in[8];
  const float* b3  = (const float*)d_in[9];
  const float* W3r = (const float*)d_in[10];
  float* out = (float*)d_out;

  const int IN_DIM = 128, HID = 256, OUT3 = 128;
  const int N = in_sizes[0] / IN_DIM;
  const int E = in_sizes[1] / 2;
  const int* src = ei;
  const int* dst = ei + E;

  auto align = [](size_t o) { return (o + 255) & ~(size_t)255; };
  char* ws = (char*)d_ws;
  size_t off = 0;
  float*          cnt = (float*)(ws + off);          off = align(off + (size_t)N * 4);
  float*          agg = (float*)(ws + off);          off = align(off + (size_t)N * HID * 4);
  unsigned int*   abf = (unsigned int*)(ws + off);   off = align(off + (size_t)N * 2 * HID * 2);
  float*          hA  = (float*)(ws + off);          off = align(off + (size_t)N * HID * 4);
  float*          hB  = (float*)(ws + off);          off = align(off + (size_t)N * HID * 4);
  unsigned short* w1  = (unsigned short*)(ws + off); off = align(off + (size_t)HID * 2 * IN_DIM * 2);
  unsigned short* w2  = (unsigned short*)(ws + off); off = align(off + (size_t)HID * 2 * HID * 2);
  unsigned short* w3  = (unsigned short*)(ws + off); off = align(off + (size_t)OUT3 * 2 * HID * 2);
  (void)ws_size; (void)n_in; (void)out_size;

  const int TB = 256;
  auto blocks = [&](long long work) { return (unsigned)((work + TB - 1) / TB); };

  // Pack weights to bf16 [Wl|Wr]
  build_wbf<<<blocks((long long)HID * 2 * IN_DIM), TB, 0, stream>>>(W1l, W1r, w1, HID, IN_DIM);
  build_wbf<<<blocks((long long)HID * 2 * HID),   TB, 0, stream>>>(W2l, W2r, w2, HID, HID);
  build_wbf<<<blocks((long long)OUT3 * 2 * HID),  TB, 0, stream>>>(W3l, W3r, w3, OUT3, HID);

  // Degree (same for all layers): cnt -> 1/max(cnt,1)
  fill_zero_f32<<<blocks(N), TB, 0, stream>>>(cnt, N);
  count_edges<<<blocks(E), TB, 0, stream>>>(dst, E, cnt);
  invert_cnt<<<blocks(N), TB, 0, stream>>>(cnt, N);

  auto layer = [&](const float* in, int din, const unsigned short* wbf, const float* bias,
                   int dout, float* o) {
    long long aggN = (long long)N * din;
    fill_zero_f32<<<blocks(aggN), TB, 0, stream>>>(agg, (int)aggN);
    scatter_edges<<<blocks((long long)E * (din >> 2)), TB, 0, stream>>>(in, src, dst, agg, E, din);
    build_abf<<<blocks((long long)N * din), TB, 0, stream>>>(agg, in, cnt, abf, N, din);
    dim3 grid((N + 31) / 32, dout / 128);
    sage_gemm_wmma<<<grid, 256, 0, stream>>>(abf, (const unsigned int*)wbf, bias, o,
                                             N, 2 * din, dout);
  };

  layer(x,  IN_DIM, w1, b1, HID,  hA);   // layer 1: 128 -> 256
  layer(hA, HID,    w2, b2, HID,  hB);   // layer 2: 256 -> 256
  layer(hB, HID,    w3, b3, OUT3, out);  // layer 3: 256 -> 128
}
